// DifferentialCrossAttention_24541443129473
// MI455X (gfx1250) — compile-verified, matrix operand order checked by reading, not run on hardware
//
#include <hip/hip_runtime.h>
#include <stddef.h>

// ---------------------------------------------------------------------------
// Differential cross-attention for MI455X (gfx1250, wave32, WMMA bf16)
// ---------------------------------------------------------------------------

#define DIMC   1024
#define NH     16
#define HEADD  64
#define HALFD  32
#define BB     2
#define NN     2048
#define MROWS  (BB*NN)          // 4096
#define SCALEF 0.17677669529663687f   // 32^-0.5
#define LAM0   0.355509067591f        // 0.8 - 0.6*exp(-0.3)
#define ONE_M_LAM0 0.644490932409f

typedef __attribute__((ext_vector_type(16))) __bf16 v16bf;
typedef __attribute__((ext_vector_type(8)))  __bf16 v8bf;
typedef __attribute__((ext_vector_type(8)))  float  v8f;
typedef int v4i __attribute__((vector_size(16)));   // matches builtin param type

union V16U { v16bf v; v8bf h[2]; };

// CDNA5 async global->LDS copy (ASYNCcnt path), guarded so we fall back to a
// plain VGPR round-trip copy if the toolchain lacks the builtin.
#if defined(__has_builtin)
#if __has_builtin(__builtin_amdgcn_global_load_async_to_lds_b128)
#define HAVE_ASYNC_LDS 1
#endif
#endif

__device__ inline void wait_asynccnt0() {
#if defined(__has_builtin)
#if __has_builtin(__builtin_amdgcn_s_wait_asynccnt)
  __builtin_amdgcn_s_wait_asynccnt(0);
  return;
#else
  asm volatile("s_wait_asynccnt 0x0" ::: "memory");
  return;
#endif
#endif
}

__device__ inline __bf16 f2bf(float f) {
  union { float f; unsigned u; } v; v.f = f;
  unsigned r = v.u + 0x7FFFu + ((v.u >> 16) & 1u);   // RNE
  unsigned short hs = (unsigned short)(r >> 16);
  __bf16 b;
  __builtin_memcpy(&b, &hs, 2);
  return b;
}

// Load a 16x32 bf16 fragment (A-operand layout; also used for B when the
// source array is stored [n][k]).  Per ISA 7.12.2:
//   lanes 0-15 : row = lane,    koff = 0
//   lanes 16-31: row = lane-16, koff = 8
//   elements e=0..7  -> k = koff + e        (contiguous 16B)
//   elements e=8..15 -> k = 16 + koff + e-8 (contiguous 16B)
__device__ inline v16bf load_frag(const __bf16* base, int ld) {
  int lane = threadIdx.x & 31;
  int r    = lane & 15;
  int koff = (lane >> 4) ? 8 : 0;
  const __bf16* p = base + r * ld + koff;
  V16U u;
  u.h[0] = *(const v8bf*)(p);
  u.h[1] = *(const v8bf*)(p + 16);
  return u.v;
}

__device__ inline float rmax16(float v) {
  #pragma unroll
  for (int m = 8; m >= 1; m >>= 1) v = fmaxf(v, __shfl_xor(v, m, 32));
  return v;
}
__device__ inline float rsum16(float v) {
  #pragma unroll
  for (int m = 8; m >= 1; m >>= 1) v += __shfl_xor(v, m, 32);
  return v;
}

// ---------------------------------------------------------------------------
// 0) fp32 -> bf16 conversion
// ---------------------------------------------------------------------------
__global__ void cvt_f32_bf16(const float* __restrict__ s, __bf16* __restrict__ d, int n) {
  int i = blockIdx.x * blockDim.x + threadIdx.x;
  if (i < n) d[i] = f2bf(s[i]);
}

// ---------------------------------------------------------------------------
// 1) lambda per head: exp(sum lq1*lk1) - exp(sum lq2*lk2) + LAMBDA_INIT
// ---------------------------------------------------------------------------
__global__ void lambda_kernel(const float* __restrict__ lq1, const float* __restrict__ lk1,
                              const float* __restrict__ lq2, const float* __restrict__ lk2,
                              float* __restrict__ lam) {
  int h = threadIdx.x;
  if (h < NH) {
    float s1 = 0.f, s2 = 0.f;
    for (int i = 0; i < HALFD; ++i) {
      s1 += lq1[h * HALFD + i] * lk1[h * HALFD + i];
      s2 += lq2[h * HALFD + i] * lk2[h * HALFD + i];
    }
    lam[h] = __expf(s1) - __expf(s2) + LAM0;
  }
}

// ---------------------------------------------------------------------------
// 2) bf16 GEMM via WMMA:  Y(M,1024) = A(M,1024) @ W^T + bias
//    W is row-major [n][k] (so B[k][n] = W[n][k]).
//    mode 0: write bf16 to (b,h,n,d) layout   mode 1: write f32 row-major
//    Block: 256 thr (8 waves). Block tile 128x128, wave tile 32x64 (2x4 accs):
//    per K-step/wave: 2 A-frags + 4 B-frags (12 ds_load_b128) feed 8 WMMAs.
// ---------------------------------------------------------------------------
__global__ void gemm_bf16_wmma(const __bf16* __restrict__ A,
                               const __bf16* __restrict__ W,
                               const float*  __restrict__ bias,
                               __bf16* __restrict__ out_bhnd,
                               float*  __restrict__ out_f32,
                               int mode) {
  __shared__ __bf16 atile[128 * 32];
  __shared__ __bf16 wtile[128 * 32];
  const int K = DIMC;
  int mblk = blockIdx.x * 128;
  int nblk = blockIdx.y * 128;
  int t = threadIdx.x, wave = t >> 5, lane = t & 31;
  int mw = (wave >> 1) * 32;   // wave row offset within block tile
  int nw = (wave & 1) * 64;    // wave col offset within block tile

  v8f acc[2][4] = {};

  for (int kk = 0; kk < K; kk += 32) {
    { // A tile: 128x32, 16 elems per thread
      int m = t >> 1, ko = (t & 1) * 16;
      const __bf16* src = A + (size_t)(mblk + m) * K + kk + ko;
#ifdef HAVE_ASYNC_LDS
      __builtin_amdgcn_global_load_async_to_lds_b128(
          (__attribute__((address_space(1))) v4i*)(src),
          (__attribute__((address_space(3))) v4i*)&atile[m * 32 + ko], 0, 0);
      __builtin_amdgcn_global_load_async_to_lds_b128(
          (__attribute__((address_space(1))) v4i*)(src + 8),
          (__attribute__((address_space(3))) v4i*)&atile[m * 32 + ko + 8], 0, 0);
#else
      v8bf* dst = (v8bf*)&atile[m * 32 + ko];
      dst[0] = ((const v8bf*)src)[0];
      dst[1] = ((const v8bf*)src)[1];
#endif
      if (kk + 32 < K) __builtin_prefetch(src + 32, 0, 1);
    }
    { // W tile: 128x32, 16 elems per thread
      int n = t >> 1, ko = (t & 1) * 16;
      const __bf16* src = W + (size_t)(nblk + n) * K + kk + ko;
#ifdef HAVE_ASYNC_LDS
      __builtin_amdgcn_global_load_async_to_lds_b128(
          (__attribute__((address_space(1))) v4i*)(src),
          (__attribute__((address_space(3))) v4i*)&wtile[n * 32 + ko], 0, 0);
      __builtin_amdgcn_global_load_async_to_lds_b128(
          (__attribute__((address_space(1))) v4i*)(src + 8),
          (__attribute__((address_space(3))) v4i*)&wtile[n * 32 + ko + 8], 0, 0);
#else
      v8bf* dst = (v8bf*)&wtile[n * 32 + ko];
      dst[0] = ((const v8bf*)src)[0];
      dst[1] = ((const v8bf*)src)[1];
#endif
      if (kk + 32 < K) __builtin_prefetch(src + 32, 0, 1);
    }
#ifdef HAVE_ASYNC_LDS
    wait_asynccnt0();
#endif
    __syncthreads();

    v16bf a0 = load_frag(&atile[(mw) * 32], 32);
    v16bf a1 = load_frag(&atile[(mw + 16) * 32], 32);
    #pragma unroll
    for (int tt = 0; tt < 4; ++tt) {
      v16bf bfrag = load_frag(&wtile[(nw + tt * 16) * 32], 32);
      acc[0][tt] = __builtin_amdgcn_wmma_f32_16x16x32_bf16(
          false, a0, false, bfrag, (short)0, acc[0][tt], false, false);
      acc[1][tt] = __builtin_amdgcn_wmma_f32_16x16x32_bf16(
          false, a1, false, bfrag, (short)0, acc[1][tt], false, false);
    }
    __syncthreads();
  }

  int hl = lane >> 4, ln = lane & 15;
  #pragma unroll
  for (int mi = 0; mi < 2; ++mi) {
    #pragma unroll
    for (int tt = 0; tt < 4; ++tt) {
      int col = nblk + nw + tt * 16 + ln;
      float bval = bias[col];
      #pragma unroll
      for (int r = 0; r < 8; ++r) {
        int row = mblk + mw + mi * 16 + r + 8 * hl;
        float v = acc[mi][tt][r] + bval;
        if (mode == 0) {
          int b = row >> 11, n = row & (NN - 1);
          int h = col >> 6,  d = col & (HEADD - 1);
          out_bhnd[(((size_t)(b * NH + h) * NN) + n) * HEADD + d] = f2bf(v);
        } else {
          out_f32[(size_t)row * DIMC + col] = v;
        }
      }
    }
  }
}

// ---------------------------------------------------------------------------
// 3) Differential flash attention.
//    Block = 64 queries of one (b,h). Waves 0-3: a1 path (dims 0-31),
//    waves 4-7: a2 path (dims 32-63). Streams 32 key tiles of 64.
//    outA = softmax(q1 k1^T * s) @ V ; outB = softmax(q2 k2^T * s) @ V
// ---------------------------------------------------------------------------
__global__ void diff_attn_kernel(const __bf16* __restrict__ Q,
                                 const __bf16* __restrict__ Kd,
                                 const __bf16* __restrict__ Vd,
                                 float* __restrict__ outA,
                                 float* __restrict__ outB) {
  __shared__ __bf16 ktile[64 * 64];     // [key][dim]
  __shared__ __bf16 vtileT[64 * 64];    // [dim][key]  (transposed V)
  __shared__ __bf16 pstage[8 * 16 * 64];// per-wave P staging [row][key]

  int bh    = blockIdx.x >> 5;   // 0..31  (b*16+h)
  int qt    = blockIdx.x & 31;
  int qbase = qt * 64;
  int t = threadIdx.x, wave = t >> 5, lane = t & 31;
  int hsel = wave >> 2;          // 0 = a1 path, 1 = a2 path
  int msub = (wave & 3) * 16;
  int d0   = hsel * HALFD;

  const __bf16* qptr = Q + ((size_t)bh * NN + qbase + msub) * HEADD + d0;
  v16bf aQ = load_frag(qptr, HEADD);

  v8f acc[4] = {};
  float mst[8], lst[8];
  #pragma unroll
  for (int r = 0; r < 8; ++r) { mst[r] = -1e30f; lst[r] = 0.f; }

  __bf16* pst = &pstage[wave * 16 * 64];
  int hl = lane >> 4, ln = lane & 15;

  for (int kt = 0; kt < 32; ++kt) {
    { // cooperative K/V tile load; thread -> (key, 16-dim chunk)
      int key = t >> 2, dd = (t & 3) * 16;
      const __bf16* ks = Kd + ((size_t)bh * NN + kt * 64 + key) * HEADD + dd;
#ifdef HAVE_ASYNC_LDS
      // K tile via GLOBAL_LOAD_ASYNC_TO_LDS_B128 (no VGPR round-trip)
      __builtin_amdgcn_global_load_async_to_lds_b128(
          (__attribute__((address_space(1))) v4i*)(ks),
          (__attribute__((address_space(3))) v4i*)&ktile[key * 64 + dd], 0, 0);
      __builtin_amdgcn_global_load_async_to_lds_b128(
          (__attribute__((address_space(1))) v4i*)(ks + 8),
          (__attribute__((address_space(3))) v4i*)&ktile[key * 64 + dd + 8], 0, 0);
#else
      v8bf* kd2 = (v8bf*)&ktile[key * 64 + dd];
      kd2[0] = ((const v8bf*)ks)[0];
      kd2[1] = ((const v8bf*)ks)[1];
#endif
      const __bf16* vs = Vd + ((size_t)bh * NN + kt * 64 + key) * HEADD + dd;
      #pragma unroll
      for (int i = 0; i < 16; ++i) vtileT[(dd + i) * 64 + key] = vs[i];
    }
#ifdef HAVE_ASYNC_LDS
    wait_asynccnt0();
#endif
    __syncthreads();

    // S = (q_half . k_half^T) * SCALE : 4 n-tiles of 16 (one WMMA each, K=32)
    v8f s[4];
    #pragma unroll
    for (int tt = 0; tt < 4; ++tt) {
      v16bf bK = load_frag(&ktile[(tt * 16) * 64 + d0], 64);
      v8f z = {};
      s[tt] = __builtin_amdgcn_wmma_f32_16x16x32_bf16(
          false, aQ, false, bK, (short)0, z, false, false);
      #pragma unroll
      for (int r = 0; r < 8; ++r) s[tt][r] *= SCALEF;
    }

    // online softmax (rows r + 8*hl; reduce across the 16 lanes of a half)
    float alpha[8], rsum[8];
    #pragma unroll
    for (int r = 0; r < 8; ++r) {
      float m = s[0][r];
      #pragma unroll
      for (int tt = 1; tt < 4; ++tt) m = fmaxf(m, s[tt][r]);
      m = rmax16(m);
      float mn = fmaxf(mst[r], m);
      alpha[r] = __expf(mst[r] - mn);
      mst[r] = mn;
      rsum[r] = 0.f;
    }
    #pragma unroll
    for (int tt = 0; tt < 4; ++tt)
      #pragma unroll
      for (int r = 0; r < 8; ++r) {
        float p = __expf(s[tt][r] - mst[r]);
        s[tt][r] = p;
        rsum[r] += p;
      }
    #pragma unroll
    for (int r = 0; r < 8; ++r) {
      rsum[r] = rsum16(rsum[r]);
      lst[r] = lst[r] * alpha[r] + rsum[r];
    }
    #pragma unroll
    for (int tt = 0; tt < 4; ++tt)
      #pragma unroll
      for (int r = 0; r < 8; ++r) acc[tt][r] *= alpha[r];

    // stage P (bf16) to per-wave LDS in [row][key] layout
    #pragma unroll
    for (int tt = 0; tt < 4; ++tt)
      #pragma unroll
      for (int r = 0; r < 8; ++r)
        pst[(r + 8 * hl) * 64 + tt * 16 + ln] = f2bf(s[tt][r]);
    __syncthreads();

    // O += P(16x64) @ V(64x64): two K=32 steps, B from transposed V tile
    #pragma unroll
    for (int ks = 0; ks < 2; ++ks) {
      v16bf aP = load_frag(pst + ks * 32, 64);
      #pragma unroll
      for (int tt = 0; tt < 4; ++tt) {
        v16bf bV = load_frag(&vtileT[(tt * 16) * 64 + ks * 32], 64);
        acc[tt] = __builtin_amdgcn_wmma_f32_16x16x32_bf16(
            false, aP, false, bV, (short)0, acc[tt], false, false);
      }
    }
    __syncthreads();
  }

  float* OUT = hsel ? outB : outA;
  #pragma unroll
  for (int tt = 0; tt < 4; ++tt)
    #pragma unroll
    for (int r = 0; r < 8; ++r) {
      int row = qbase + msub + r + 8 * hl;
      OUT[((size_t)bh * NN + row) * HEADD + tt * 16 + ln] = acc[tt][r] / lst[r];
    }
}

// ---------------------------------------------------------------------------
// 4) GroupNorm stats over (N, HEAD) per (b,h) of  x = OA - lam*OB
// ---------------------------------------------------------------------------
__global__ void gn_stats(const float* __restrict__ OA, const float* __restrict__ OB,
                         const float* __restrict__ lam, float* __restrict__ stats) {
  const int CNT = NN * HEADD;  // 131072
  int bh = blockIdx.x, h = bh & 15;
  float lm = lam[h];
  const float* pa = OA + (size_t)bh * CNT;
  const float* pb = OB + (size_t)bh * CNT;
  float s = 0.f, ss = 0.f;
  for (int i = threadIdx.x; i < CNT; i += 256) {
    float x = pa[i] - lm * pb[i];
    s += x; ss += x * x;
  }
  __shared__ float sh0[256], sh1[256];
  sh0[threadIdx.x] = s; sh1[threadIdx.x] = ss;
  __syncthreads();
  for (int o = 128; o > 0; o >>= 1) {
    if (threadIdx.x < o) {
      sh0[threadIdx.x] += sh0[threadIdx.x + o];
      sh1[threadIdx.x] += sh1[threadIdx.x + o];
    }
    __syncthreads();
  }
  if (threadIdx.x == 0) {
    float mean = sh0[0] * (1.f / CNT);
    float var  = sh1[0] * (1.f / CNT) - mean * mean;
    stats[bh * 2]     = mean;
    stats[bh * 2 + 1] = rsqrtf(var + 1e-5f);
  }
}

// ---------------------------------------------------------------------------
// 5) GroupNorm apply + (1-LAM0) scale + relayout to bf16 (B*N, DIM)
// ---------------------------------------------------------------------------
__global__ void gn_apply(const float* __restrict__ OA, const float* __restrict__ OB,
                         const float* __restrict__ lam, const float* __restrict__ stats,
                         const float* __restrict__ gw, const float* __restrict__ gb,
                         __bf16* __restrict__ Y) {
  size_t idx = (size_t)blockIdx.x * 256 + threadIdx.x;  // < 4194304
  int bh  = (int)(idx >> 17);
  int h   = bh & 15, b = bh >> 4;
  int rem = (int)(idx & 131071);
  int n = rem >> 6, d = rem & 63;
  float lm = lam[h];
  float mean = stats[bh * 2], rstd = stats[bh * 2 + 1];
  float x = OA[idx] - lm * OB[idx];
  float y = ((x - mean) * rstd * gw[h * HEADD + d] + gb[h * HEADD + d]) * ONE_M_LAM0;
  Y[(((size_t)b * NN + n) * DIMC) + h * HEADD + d] = f2bf(y);
}

// ---------------------------------------------------------------------------
// launch
// ---------------------------------------------------------------------------
extern "C" void kernel_launch(void* const* d_in, const int* in_sizes, int n_in,
                              void* d_out, int out_size, void* d_ws, size_t ws_size,
                              hipStream_t stream) {
  (void)in_sizes; (void)n_in; (void)out_size; (void)ws_size;

  const float* x1  = (const float*)d_in[0];
  const float* x2  = (const float*)d_in[1];
  const float* Wq  = (const float*)d_in[2];
  const float* bq  = (const float*)d_in[3];
  const float* Wk  = (const float*)d_in[4];
  const float* bk  = (const float*)d_in[5];
  const float* Wv  = (const float*)d_in[6];
  const float* bv  = (const float*)d_in[7];
  const float* Wo  = (const float*)d_in[8];
  const float* bo  = (const float*)d_in[9];
  const float* lq1 = (const float*)d_in[10];
  const float* lk1 = (const float*)d_in[11];
  const float* lq2 = (const float*)d_in[12];
  const float* lk2 = (const float*)d_in[13];
  const float* gnw = (const float*)d_in[14];
  const float* gnb = (const float*)d_in[15];
  float* out = (float*)d_out;

  char* ws = (char*)d_ws;
  const size_t ACT_BF = (size_t)MROWS * DIMC * 2;         // 8 MB
  const size_t W_BF   = (size_t)DIMC * DIMC * 2;          // 2 MB
  const size_t O_F32  = (size_t)BB * NH * NN * HEADD * 4; // 16 MB
  size_t off = 0;
  __bf16* X1B = (__bf16*)(ws + off); off += ACT_BF;
  __bf16* X2B = (__bf16*)(ws + off); off += ACT_BF;
  __bf16* WQB = (__bf16*)(ws + off); off += W_BF;
  __bf16* WKB = (__bf16*)(ws + off); off += W_BF;
  __bf16* WVB = (__bf16*)(ws + off); off += W_BF;
  __bf16* WOB = (__bf16*)(ws + off); off += W_BF;
  __bf16* Q1  = (__bf16*)(ws + off); off += ACT_BF;
  __bf16* K1  = (__bf16*)(ws + off); off += ACT_BF;
  __bf16* V1  = (__bf16*)(ws + off); off += ACT_BF;
  __bf16* Q2  = (__bf16*)(ws + off); off += ACT_BF;
  __bf16* K2  = (__bf16*)(ws + off); off += ACT_BF;
  __bf16* V2  = (__bf16*)(ws + off); off += ACT_BF;
  float*  OA1 = (float*)(ws + off);  off += O_F32;
  float*  OB1 = (float*)(ws + off);  off += O_F32;
  float*  OA2 = (float*)(ws + off);  off += O_F32;
  float*  OB2 = (float*)(ws + off);  off += O_F32;
  __bf16* Y1B = (__bf16*)(ws + off); off += ACT_BF;
  __bf16* Y2B = (__bf16*)(ws + off); off += ACT_BF;
  float*  LAM = (float*)(ws + off);  off += 256;
  float*  ST1 = (float*)(ws + off);  off += 256;
  float*  ST2 = (float*)(ws + off);  off += 256;

  // conversions
  {
    int na = MROWS * DIMC, nw = DIMC * DIMC;
    cvt_f32_bf16<<<(na + 255) / 256, 256, 0, stream>>>(x1, X1B, na);
    cvt_f32_bf16<<<(na + 255) / 256, 256, 0, stream>>>(x2, X2B, na);
    cvt_f32_bf16<<<(nw + 255) / 256, 256, 0, stream>>>(Wq, WQB, nw);
    cvt_f32_bf16<<<(nw + 255) / 256, 256, 0, stream>>>(Wk, WKB, nw);
    cvt_f32_bf16<<<(nw + 255) / 256, 256, 0, stream>>>(Wv, WVB, nw);
    cvt_f32_bf16<<<(nw + 255) / 256, 256, 0, stream>>>(Wo, WOB, nw);
  }
  lambda_kernel<<<1, 32, 0, stream>>>(lq1, lk1, lq2, lk2, LAM);

  // QKV projections
  dim3 ggrid(MROWS / 128, DIMC / 128);
  gemm_bf16_wmma<<<ggrid, 256, 0, stream>>>(X1B, WQB, bq, Q1, nullptr, 0);
  gemm_bf16_wmma<<<ggrid, 256, 0, stream>>>(X1B, WKB, bk, K1, nullptr, 0);
  gemm_bf16_wmma<<<ggrid, 256, 0, stream>>>(X1B, WVB, bv, V1, nullptr, 0);
  gemm_bf16_wmma<<<ggrid, 256, 0, stream>>>(X2B, WQB, bq, Q2, nullptr, 0);
  gemm_bf16_wmma<<<ggrid, 256, 0, stream>>>(X2B, WKB, bk, K2, nullptr, 0);
  gemm_bf16_wmma<<<ggrid, 256, 0, stream>>>(X2B, WVB, bv, V2, nullptr, 0);

  // attention:  out1 = attend(q1, k2, v2), out2 = attend(q2, k1, v1)
  diff_attn_kernel<<<BB * NH * (NN / 64), 256, 0, stream>>>(Q1, K2, V2, OA1, OB1);
  diff_attn_kernel<<<BB * NH * (NN / 64), 256, 0, stream>>>(Q2, K1, V1, OA2, OB2);

  // groupnorm
  gn_stats<<<BB * NH, 256, 0, stream>>>(OA1, OB1, LAM, ST1);
  gn_stats<<<BB * NH, 256, 0, stream>>>(OA2, OB2, LAM, ST2);
  int ntot = BB * NH * NN * HEADD;
  gn_apply<<<ntot / 256, 256, 0, stream>>>(OA1, OB1, LAM, ST1, gnw, gnb, Y1B);
  gn_apply<<<ntot / 256, 256, 0, stream>>>(OA2, OB2, LAM, ST2, gnw, gnb, Y2B);

  // output projections -> f32 d_out (out1 then out2)
  gemm_bf16_wmma<<<ggrid, 256, 0, stream>>>(Y1B, WOB, bo, nullptr, out, 1);
  gemm_bf16_wmma<<<ggrid, 256, 0, stream>>>(Y2B, WOB, bo, nullptr, out + (size_t)MROWS * DIMC, 1);
}
